// WormTransformer_30064771072822
// MI455X (gfx1250) — compile-verified, matrix operand order checked by reading, not run on hardware
//
#include <hip/hip_runtime.h>
#include <hip/hip_bf16.h>

#define T_SEQ 1600
#define BATCH 32
#define NTOK (BATCH * T_SEQ)   // 51200 (b,t) tokens
#define NELEM (NTOK * 2)       // 102400 floats in x

typedef __attribute__((ext_vector_type(16))) _Float16 v16h;
typedef __attribute__((ext_vector_type(8)))  float    v8f;

// ---------------------------------------------------------------------------
// Kernel 0: x = (input + pos) * mask ; then qkv for layer 0
// ---------------------------------------------------------------------------
__global__ void k_embed_qkv(const float* __restrict__ in_emb,
                            const float* __restrict__ pos,
                            const int*   __restrict__ mask_idx,
                            const float* __restrict__ Wq,
                            const float* __restrict__ Wk,
                            const float* __restrict__ Wv,
                            float* __restrict__ x,
                            float* __restrict__ q,
                            float* __restrict__ k,
                            float* __restrict__ v) {
    int i = blockIdx.x * blockDim.x + threadIdx.x;   // token index over B*T
    if (i >= NTOK) return;
    int t = i % T_SEQ;
    int mi = *mask_idx;
    float x0 = in_emb[2*i]   + pos[2*t];
    float x1 = in_emb[2*i+1] + pos[2*t+1];
    if (mi == 0) x0 = 0.f;
    if (mi == 1) x1 = 0.f;
    x[2*i] = x0; x[2*i+1] = x1;
    // Wq/Wk/Wv layer-0 block: shape (H=2, C=2), W[h][c] at h*2+c
    q[2*i]   = Wq[0]*x0 + Wq[1]*x1;  q[2*i+1] = Wq[2]*x0 + Wq[3]*x1;
    k[2*i]   = Wk[0]*x0 + Wk[1]*x1;  k[2*i+1] = Wk[2]*x0 + Wk[3]*x1;
    v[2*i]   = Wv[0]*x0 + Wv[1]*x1;  v[2*i+1] = Wv[2]*x0 + Wv[3]*x1;
}

// ---------------------------------------------------------------------------
// Attention: one wave (32 lanes) per 16-row query tile per batch.
// QK^T via v_wmma_f32_16x16x32_f16 (K padded 2->32 with zeros).
// Scores are produced directly in the exp2 domain (log2e folded into q),
// so softmax needs only a bare v_exp_f32 per cell.
// Two-pass flash softmax: pass 1 = row max, pass 2 = exp2-sum, no rescaling.
// All tile loads are branchless (hi lanes load valid dup addresses, operand
// zeroed by select) so the unroller can pipeline loads ahead of the WMMAs.
// D layout: VGPR r, lanes 0-15 -> M=r, N=lane ; lanes 16-31 -> M=8+r, N=lane-16
// ---------------------------------------------------------------------------
__global__ __launch_bounds__(32)
void k_attn(const float* __restrict__ q,
            const float* __restrict__ kk,
            const float* __restrict__ vv,
            float* __restrict__ x) {
    const int lane = threadIdx.x;          // 0..31
    const int qt   = blockIdx.x;           // 0..99  query tile
    const int b    = blockIdx.y;           // 0..31  batch
    const int base = b * T_SEQ * 2;
    const bool lo  = (lane < 16);
    const int l15  = lane & 15;

    const float2* __restrict__ qp = (const float2*)(q  + base);
    const float2* __restrict__ kp = (const float2*)(kk + base);
    const float2* __restrict__ vp = (const float2*)(vv + base);

    // A operand: 16x32 f16, only K=0,1 nonzero (lanes 0-15, elems 0,1).
    // Scale = rsqrt(head_size) * log2(e): scores exit the WMMA in exp2 domain.
    v16h A = {};
    {
        const float sc = 0.70710678118654752f * 1.44269504088896340f;
        float2 qv = qp[qt * 16 + l15];     // all lanes load a valid address
        A[0] = (_Float16)(lo ? qv.x * sc : 0.f);
        A[1] = (_Float16)(lo ? qv.y * sc : 0.f);
    }

    // ---- Pass 1: row maxima (distributed per lane, merged after) ----------
    float m[8];
#pragma unroll
    for (int r = 0; r < 8; ++r) m[r] = -1e30f;

    for (int kt = 0; kt < T_SEQ / 16; ++kt) {
        float2 kv = kp[kt * 16 + l15];     // branchless: all lanes load
        v16h Bv = {};
        Bv[0] = (_Float16)(lo ? kv.x : 0.f);
        Bv[1] = (_Float16)(lo ? kv.y : 0.f);
        v8f c = {};
        c = __builtin_amdgcn_wmma_f32_16x16x32_f16(
                false, A, false, Bv, (short)0, c, false, false);
#pragma unroll
        for (int r = 0; r < 8; ++r) m[r] = fmaxf(m[r], c[r]);
    }

    // Merge maxima within each 16-lane group -> true row max in every lane.
#pragma unroll
    for (int r = 0; r < 8; ++r) {
#pragma unroll
        for (int msk = 1; msk < 16; msk <<= 1)
            m[r] = fmaxf(m[r], __shfl_xor(m[r], msk, 32));
    }

    // ---- Pass 2: exp2-sums and weighted-v numerators, no rescaling --------
    float d[8], n0[8], n1[8];
#pragma unroll
    for (int r = 0; r < 8; ++r) { d[r] = 0.f; n0[r] = 0.f; n1[r] = 0.f; }

    for (int kt = 0; kt < T_SEQ / 16; ++kt) {
        float2 kv = kp[kt * 16 + l15];
        float2 vvv = vp[kt * 16 + l15];    // all 8 scores share this column
        v16h Bv = {};
        Bv[0] = (_Float16)(lo ? kv.x : 0.f);
        Bv[1] = (_Float16)(lo ? kv.y : 0.f);
        v8f c = {};
        c = __builtin_amdgcn_wmma_f32_16x16x32_f16(
                false, A, false, Bv, (short)0, c, false, false);
#pragma unroll
        for (int r = 0; r < 8; ++r) {
            float p = exp2f(c[r] - m[r]);  // == exp(score - max), <= 1
            d[r]  += p;
            n0[r] += p * vvv.x;
            n1[r] += p * vvv.y;
        }
    }

    // Merge partial sums (plain adds; all partials share the same row max).
#pragma unroll
    for (int r = 0; r < 8; ++r) {
#pragma unroll
        for (int msk = 1; msk < 16; msk <<= 1) {
            d[r]  += __shfl_xor(d[r],  msk, 32);
            n0[r] += __shfl_xor(n0[r], msk, 32);
            n1[r] += __shfl_xor(n1[r], msk, 32);
        }
    }

    // Lane 0 writes rows M=0..7, lane 16 writes rows M=8..15 (residual add).
    if (l15 == 0) {
#pragma unroll
        for (int r = 0; r < 8; ++r) {
            int M = r + (lane >> 4) * 8;
            int t = qt * 16 + M;
            float inv = 1.0f / d[r];
            x[base + 2*t]     += n0[r] * inv;
            x[base + 2*t + 1] += n1[r] * inv;
        }
    }
}

// ---------------------------------------------------------------------------
// LN2 + FFN(2->8->relu->2) + residual; then qkv for next layer if any.
// ---------------------------------------------------------------------------
__global__ void k_ffn_qkv(const float* __restrict__ W1, const float* __restrict__ b1,
                          const float* __restrict__ W2, const float* __restrict__ b2,
                          const float* __restrict__ g,  const float* __restrict__ bb,
                          const float* __restrict__ Wq, const float* __restrict__ Wk,
                          const float* __restrict__ Wv,
                          float* __restrict__ x,
                          float* __restrict__ q, float* __restrict__ k, float* __restrict__ v,
                          int has_next) {
    int i = blockIdx.x * blockDim.x + threadIdx.x;
    if (i >= NTOK) return;
    float x0 = x[2*i], x1 = x[2*i+1];

    float mu = 0.5f * (x0 + x1);
    float d0 = x0 - mu, d1 = x1 - mu;
    float var = 0.5f * (d0*d0 + d1*d1);
    float rs = rsqrtf(var + 1e-5f);
    float h0 = d0 * rs * g[0] + bb[0];
    float h1 = d1 * rs * g[1] + bb[1];

    float acc0 = b2[0], acc1 = b2[1];
#pragma unroll
    for (int j = 0; j < 8; ++j) {
        float hj = fmaxf(W1[2*j]*h0 + W1[2*j+1]*h1 + b1[j], 0.f);  // W1 (8,2)
        acc0 += W2[j]     * hj;                                    // W2 (2,8)
        acc1 += W2[8 + j] * hj;
    }
    x0 += acc0; x1 += acc1;
    x[2*i] = x0; x[2*i+1] = x1;

    if (has_next) {
        q[2*i] = Wq[0]*x0 + Wq[1]*x1;  q[2*i+1] = Wq[2]*x0 + Wq[3]*x1;
        k[2*i] = Wk[0]*x0 + Wk[1]*x1;  k[2*i+1] = Wk[2]*x0 + Wk[3]*x1;
        v[2*i] = Wv[0]*x0 + Wv[1]*x1;  v[2*i+1] = Wv[2]*x0 + Wv[3]*x1;
    }
}

// ---------------------------------------------------------------------------
// Loss: copy y to d_out and deterministic two-pass MSE reduction.
// ---------------------------------------------------------------------------
__global__ void k_loss_partial(const float* __restrict__ x,
                               const float* __restrict__ tgt,
                               float* __restrict__ out_y,
                               float* __restrict__ part) {
    __shared__ float sdata[256];
    float s = 0.f;
    for (int i = blockIdx.x * 256 + threadIdx.x; i < NELEM; i += 256 * 256) {
        float yv = x[i];
        out_y[i] = yv;
        float df = yv - tgt[i];
        s += df * df;
    }
    sdata[threadIdx.x] = s;
    __syncthreads();
    for (int ofs = 128; ofs > 0; ofs >>= 1) {
        if (threadIdx.x < ofs) sdata[threadIdx.x] += sdata[threadIdx.x + ofs];
        __syncthreads();
    }
    if (threadIdx.x == 0) part[blockIdx.x] = sdata[0];
}

__global__ void k_loss_final(const float* __restrict__ part,
                             float* __restrict__ out_loss) {
    __shared__ float sdata[256];
    sdata[threadIdx.x] = part[threadIdx.x];   // exactly 256 partials
    __syncthreads();
    for (int ofs = 128; ofs > 0; ofs >>= 1) {
        if (threadIdx.x < ofs) sdata[threadIdx.x] += sdata[threadIdx.x + ofs];
        __syncthreads();
    }
    if (threadIdx.x == 0) *out_loss = sdata[0] * (1.0f / (float)NELEM);
}

// ---------------------------------------------------------------------------
extern "C" void kernel_launch(void* const* d_in, const int* in_sizes, int n_in,
                              void* d_out, int out_size, void* d_ws, size_t ws_size,
                              hipStream_t stream) {
    const float* in_emb = (const float*)d_in[0];
    const float* tgt    = (const float*)d_in[1];
    const int*   midx   = (const int*)  d_in[2];
    const float* pos    = (const float*)d_in[3];
    const float* Wq     = (const float*)d_in[4];   // (3,2,2)
    const float* Wk     = (const float*)d_in[5];
    const float* Wv     = (const float*)d_in[6];
    const float* W1     = (const float*)d_in[7];   // (3,8,2)
    const float* b1     = (const float*)d_in[8];   // (3,8)
    const float* W2     = (const float*)d_in[9];   // (3,2,8)
    const float* b2     = (const float*)d_in[10];  // (3,2)
    const float* g      = (const float*)d_in[11];  // (3,2)
    const float* bb     = (const float*)d_in[12];  // (3,2)

    float* ws   = (float*)d_ws;
    float* x    = ws;
    float* q    = ws + NELEM;
    float* k    = ws + 2 * NELEM;
    float* v    = ws + 3 * NELEM;
    float* part = ws + 4 * NELEM;   // 256 floats

    k_embed_qkv<<<(NTOK + 255) / 256, 256, 0, stream>>>(
        in_emb, pos, midx, Wq, Wk, Wv, x, q, k, v);

    for (int l = 0; l < 3; ++l) {
        dim3 ag(T_SEQ / 16, BATCH);
        k_attn<<<ag, 32, 0, stream>>>(q, k, v, x);
        int has_next = (l < 2) ? 1 : 0;
        // next-layer weight pointers (never dereferenced when has_next==0)
        int nl = (l < 2) ? (l + 1) : l;
        k_ffn_qkv<<<(NTOK + 255) / 256, 256, 0, stream>>>(
            W1 + l * 16, b1 + l * 8, W2 + l * 16, b2 + l * 2,
            g + l * 2, bb + l * 2,
            Wq + nl * 4, Wk + nl * 4, Wv + nl * 4,
            x, q, k, v, has_next);
    }

    float* out_y = (float*)d_out;
    k_loss_partial<<<256, 256, 0, stream>>>(x, tgt, out_y, part);
    k_loss_final<<<1, 256, 0, stream>>>(part, out_y + NELEM);
}
